// KtModule_29411936043663
// MI455X (gfx1250) — compile-verified, matrix operand order checked by reading, not run on hardware
//
#include <hip/hip_runtime.h>
#include <hip/hip_bf16.h>
#include <math.h>

typedef __attribute__((ext_vector_type(16))) _Float16 v16h;
typedef __attribute__((ext_vector_type(8)))  float    v8f;

#define DEV __device__ __forceinline__

DEV float frcp(float x)      { return __builtin_amdgcn_rcpf(x); }        // v_rcp_f32
DEV float fast_sig(float x)  { return frcp(1.0f + __expf(-x)); }
DEV float fast_silu(float x) { return x * fast_sig(x); }
DEV float reluf(float x)     { return fmaxf(x, 0.0f); }
DEV float fast_tanh(float x) {           // stable for all x, ~1 ulp
  const float ax = fabsf(x);
  const float e  = __expf(-2.0f * ax);
  const float r  = (1.0f - e) * frcp(1.0f + e);
  return copysignf(r, x);
}

// Broadcast from a compile-time lane: v_readlane_b32 -> SGPR (no LDS trip).
DEV float rlane(float v, int l) {
  return __int_as_float(__builtin_amdgcn_readlane(__float_as_int(v), l));
}
DEV void bcast4(float v[4], float s) {
  v[0] = rlane(s, 0); v[1] = rlane(s, 1); v[2] = rlane(s, 2); v[3] = rlane(s, 3);
}

// ---------------------------------------------------------------------------
// Kernel 1: A[t][n] = relu(x[t][0]*W1[n][0] + x[t][1]*W1[n][1] + b1[n])
// T x 24 via v_wmma_f32_16x16x32_f16; fp32 accuracy recovered with an f16
// hi/lo split packed into spare K slots (K0,1: xh.wh  K2,3: xl.wh  K4,5: xh.wl).
// ---------------------------------------------------------------------------
__global__ void prep_fc1_wmma(const float* __restrict__ x,
                              const float* __restrict__ W1,
                              const float* __restrict__ b1,
                              float* __restrict__ Aout, int T) {
  const int lane = threadIdx.x & 31;
  const int wave = (int)((blockIdx.x * blockDim.x + threadIdx.x) >> 5);
  const int ntiles = (T + 15) >> 4;
  if (wave >= ntiles) return;
  const int rowbase = wave << 4;
  const int m = lane & 15;
  const bool lo = lane < 16;

  float x0 = 0.f, x1 = 0.f;
  const int row = rowbase + m;
  if (lo && row < T) { x0 = x[row * 2 + 0]; x1 = x[row * 2 + 1]; }
  _Float16 x0h = (_Float16)x0, x1h = (_Float16)x1;
  _Float16 x0l = (_Float16)(x0 - (float)x0h);
  _Float16 x1l = (_Float16)(x1 - (float)x1h);

  v16h A;
#pragma unroll
  for (int i = 0; i < 16; ++i) A[i] = (_Float16)0.f;
  if (lo) { A[0] = x0h; A[1] = x1h; A[2] = x0l; A[3] = x1l; A[4] = x0h; A[5] = x1h; }

  const bool fulltile = (rowbase + 16 <= T);

#pragma unroll
  for (int nt = 0; nt < 2; ++nt) {
    const int n = nt * 16 + m;
    const bool nvalid = (n < 24);

    float w0 = 0.f, w1 = 0.f;
    if (lo && nvalid) { w0 = W1[n * 2 + 0]; w1 = W1[n * 2 + 1]; }
    _Float16 w0h = (_Float16)w0, w1h = (_Float16)w1;
    _Float16 w0l = (_Float16)(w0 - (float)w0h);
    _Float16 w1l = (_Float16)(w1 - (float)w1h);
    v16h B;
#pragma unroll
    for (int i = 0; i < 16; ++i) B[i] = (_Float16)0.f;
    if (lo) { B[0] = w0h; B[1] = w1h; B[2] = w0h; B[3] = w1h; B[4] = w0l; B[5] = w1l; }

    const float bn = nvalid ? b1[n] : 0.f;
    v8f C;
#pragma unroll
    for (int r = 0; r < 8; ++r) C[r] = bn;

    v8f D = __builtin_amdgcn_wmma_f32_16x16x32_f16(
        false, A, false, B, (short)0, C, false, false);

    const int mbase = rowbase + (lo ? 0 : 8);
    if (nvalid) {                    // single per-lane predicate around stores
      if (fulltile) {
#pragma unroll
        for (int r = 0; r < 8; ++r) Aout[(size_t)(mbase + r) * 24 + n] = reluf(D[r]);
      } else {
#pragma unroll
        for (int r = 0; r < 8; ++r)
          if (mbase + r < T) Aout[(size_t)(mbase + r) * 24 + n] = reluf(D[r]);
      }
    }
  }
}

// ---------------------------------------------------------------------------
// Kernel 2: sequential scan, one wave. Lane l owns channel j=l&3; every
// broadcast is v_readlane (SGPR), LSTM gates are computed locally (no
// shuffles). a_t loads are double-buffered one step ahead.
// ---------------------------------------------------------------------------
DEV void load_a24(float a[24], const float* __restrict__ p) {
  const float4* ap = (const float4*)p;   // 96B offset -> 16B aligned
#pragma unroll
  for (int q = 0; q < 6; ++q) {
    float4 v = ap[q];
    a[4*q+0] = v.x; a[4*q+1] = v.y; a[4*q+2] = v.z; a[4*q+3] = v.w;
  }
}

__global__ __launch_bounds__(32, 1)
void scan_kernel(const float* __restrict__ x,
                 const float* __restrict__ W1,   const float* __restrict__ b1,
                 const float* __restrict__ cv1w, const float* __restrict__ cv1b,
                 const float* __restrict__ cv2w, const float* __restrict__ cv2b,
                 const float* __restrict__ cv3w, const float* __restrict__ cv3b,
                 const float* __restrict__ cv4w, const float* __restrict__ cv4b,
                 const float* __restrict__ cv5w, const float* __restrict__ cv5b,
                 const float* __restrict__ cv6w, const float* __restrict__ cv6b,
                 const float* __restrict__ cv7w, const float* __restrict__ cv7b,
                 const float* __restrict__ Wih,  const float* __restrict__ bih,
                 const float* __restrict__ bhh,
                 const float* __restrict__ W3,   const float* __restrict__ b3,
                 const float* __restrict__ W4,   const float* __restrict__ b4,
                 const float* __restrict__ W5,   const float* __restrict__ b5,
                 const float* __restrict__ Aprec,
                 float* __restrict__ out, int T) {
  const int lane = threadIdx.x & 31;
  const int j = lane & 3;              // channel owned by this lane

  // ---- per-lane resident weights (u = [0(4), a(24), P(4)]) ----
  float wc1a[24], wc2a[24], wc1p[4], wc2p[4];
#pragma unroll
  for (int i = 0; i < 24; ++i) { wc1a[i] = cv1w[j*32 + 4 + i]; wc2a[i] = cv2w[j*32 + 4 + i]; }
#pragma unroll
  for (int i = 0; i < 4;  ++i) { wc1p[i] = cv1w[j*32 + 28 + i]; wc2p[i] = cv2w[j*32 + 28 + i]; }
  float wc3[4], wc4[4], wc6[4];
#pragma unroll
  for (int i = 0; i < 4;  ++i) { wc3[i] = cv3w[j*4+i]; wc4[i] = cv4w[j*4+i]; wc6[i] = cv6w[j*4+i]; }
  float wc5[16], w3r[16];
#pragma unroll
  for (int i = 0; i < 16; ++i) { wc5[i] = cv5w[j*16+i]; w3r[i] = W3[j*16+i]; }
  float wc7[8], w4r[8], w5r[8];
#pragma unroll
  for (int i = 0; i < 8;  ++i) { wc7[i] = cv7w[j*8+i]; w4r[i] = W4[j*8+i]; w5r[i] = W5[j*8+i]; }
  // LSTM: lane owns rows j (i-gate), 8+j (g-gate), 12+j (o-gate); f2[4:8]==0.
  float wii[4], wgg[4], woo[4];
#pragma unroll
  for (int i = 0; i < 4;  ++i) {
    wii[i] = Wih[j*8 + i];
    wgg[i] = Wih[(8 + j)*8 + i];
    woo[i] = Wih[(12 + j)*8 + i];
  }
  const float bgi = bih[j]      + bhh[j];
  const float bgg = bih[8 + j]  + bhh[8 + j];
  const float bgo = bih[12 + j] + bhh[12 + j];
  const float bc1 = cv1b[j], bc2 = cv2b[j], bc3 = cv3b[j], bc4 = cv4b[j];
  const float bc5 = cv5b[j], bc6 = cv6b[j], bc7 = cv7b[j];
  const float b3v = b3[j], b4v = b4[j], b5v = b5[j];

  const bool useA = (Aprec != nullptr);
  const float inv_s2 = 0.70710678118654752440f;   // 1/sqrt(2)
  const bool jb0 = (j & 1) != 0, jb1 = (j & 2) != 0;

  float P[4] = {0.f, 0.f, 0.f, 0.f};

  auto compute_a = [&](float a[24], int t) {
    const float xt0 = x[t*2 + 0], xt1 = x[t*2 + 1];
#pragma unroll
    for (int i = 0; i < 24; ++i)
      a[i] = reluf(fmaf(W1[2*i], xt0, fmaf(W1[2*i+1], xt1, b1[i])));
  };

  auto step = [&](const float a[24], int t) {
    // ---- x1 = silu(cv1@u+b), y2 = silu(cv2@u+b): split accumulators ----
    float s1a = bc1, s1b = 0.f, s2a = bc2, s2b = 0.f;
#pragma unroll
    for (int i = 0; i < 12; ++i) { s1a = fmaf(wc1a[i], a[i], s1a); s2a = fmaf(wc2a[i], a[i], s2a); }
#pragma unroll
    for (int i = 12; i < 24; ++i){ s1b = fmaf(wc1a[i], a[i], s1b); s2b = fmaf(wc2a[i], a[i], s2b); }
#pragma unroll
    for (int i = 0; i < 4;  ++i) { s1a = fmaf(wc1p[i], P[i], s1a); s2a = fmaf(wc2p[i], P[i], s2a); }
    float x1v[4], y2v[4];
    bcast4(x1v, fast_silu(s1a + s1b));
    bcast4(y2v, fast_silu(s2a + s2b));

    // ---- x3, x4 ----
    float s3 = bc3;
#pragma unroll
    for (int i = 0; i < 4; ++i) s3 = fmaf(wc3[i], x1v[i], s3);
    float x3v[4]; bcast4(x3v, fast_silu(s3));
    float s4 = bc4;
#pragma unroll
    for (int i = 0; i < 4; ++i) s4 = fmaf(wc4[i], x3v[i], s4);
    float x4v[4]; bcast4(x4v, fast_silu(s4));

    // ---- cat16 = [x4, r, r, r]; x5 = silu(cv5@cat16 + b5) ----
    float s5 = bc5;
#pragma unroll
    for (int i = 0; i < 4; ++i) {
      const float rr = reluf(x4v[i]);
      s5 = fmaf(wc5[i], x4v[i], s5);
      s5 = fmaf(wc5[4+i] + wc5[8+i] + wc5[12+i], rr, s5);
    }
    float x5v[4]; bcast4(x5v, fast_silu(s5));

    // ---- y1 = silu(cv6@x5+b); P1 = silu(cv7@[y1,y2]+b) ----
    float s6 = bc6;
#pragma unroll
    for (int i = 0; i < 4; ++i) s6 = fmaf(wc6[i], x5v[i], s6);
    float y1v[4]; bcast4(y1v, fast_silu(s6));
    float s7 = bc7;
#pragma unroll
    for (int i = 0; i < 4; ++i) { s7 = fmaf(wc7[i], y1v[i], s7); s7 = fmaf(wc7[4+i], y2v[i], s7); }
    float P1[4]; bcast4(P1, fast_silu(s7));

    // ---- LSTM step from zero state (per-lane gate rows, no shuffles) ----
    float gi = bgi, gg = bgg, go = bgo;
#pragma unroll
    for (int i = 0; i < 4; ++i) {
      gi = fmaf(wii[i], P1[i], gi);
      gg = fmaf(wgg[i], P1[i], gg);
      go = fmaf(woo[i], P1[i], go);
    }
    const float ccell = fast_sig(gi) * fast_tanh(gg);
    const float S0j   = fast_sig(go) * fast_tanh(ccell);
    float s0[4]; bcast4(s0, S0j);

    // ---- chan_att(S0, planes=2) ----
    const float g00 = (s0[0]*s0[0] + s0[1]*s0[1]) * inv_s2;
    const float g01 = (s0[0]*s0[2] + s0[1]*s0[3]) * inv_s2;
    const float g11 = (s0[2]*s0[2] + s0[3]*s0[3]) * inv_s2;
    float S[4];
    { const float mx = fmaxf(g00, g01);
      const float e0 = __expf(g00 - mx), e1 = __expf(g01 - mx);
      const float inv = frcp(e0 + e1); S[0] = e0 * inv; S[1] = e1 * inv; }
    { const float mx = fmaxf(g01, g11);
      const float e0 = __expf(g01 - mx), e1 = __expf(g11 - mx);
      const float inv = frcp(e0 + e1); S[2] = e0 * inv; S[3] = e1 * inv; }

    // ---- chan_att([P1,S], planes=4): lane computes row j ----
    // m rows: m0=(P1[0],P1[1]) m1=(P1[2],P1[3]) m2=(S[0],S[1]) m3=(S[2],S[3])
    const float mj0 = jb1 ? (jb0 ? S[2]  : S[0])  : (jb0 ? P1[2] : P1[0]);
    const float mj1 = jb1 ? (jb0 ? S[3]  : S[1])  : (jb0 ? P1[3] : P1[1]);
    float Gr[4];
    Gr[0] = (mj0*P1[0] + mj1*P1[1]) * inv_s2;
    Gr[1] = (mj0*P1[2] + mj1*P1[3]) * inv_s2;
    Gr[2] = (mj0*S[0]  + mj1*S[1])  * inv_s2;
    Gr[3] = (mj0*S[2]  + mj1*S[3])  * inv_s2;
    const float mx4 = fmaxf(fmaxf(Gr[0], Gr[1]), fmaxf(Gr[2], Gr[3]));
    float k16r[4];
    float esum = 0.f;
#pragma unroll
    for (int k = 0; k < 4; ++k) { k16r[k] = __expf(Gr[k] - mx4); esum += k16r[k]; }
    const float einv = frcp(esum);
#pragma unroll
    for (int k = 0; k < 4; ++k) k16r[k] *= einv;
    float k16[16];
#pragma unroll
    for (int rr = 0; rr < 4; ++rr)
#pragma unroll
      for (int cci = 0; cci < 4; ++cci) k16[4*rr + cci] = rlane(k16r[cci], rr);

    // ---- Kt = relu(W3@K16+b3) ----
    float k0 = 0.f, k1 = 0.f, k2 = 0.f, k3 = 0.f;
#pragma unroll
    for (int i = 0; i < 4; ++i) {
      k0 = fmaf(w3r[i],    k16[i],    k0);
      k1 = fmaf(w3r[4+i],  k16[4+i],  k1);
      k2 = fmaf(w3r[8+i],  k16[8+i],  k2);
      k3 = fmaf(w3r[12+i], k16[12+i], k3);
    }
    const float Ktj = reluf(b3v + ((k0 + k1) + (k2 + k3)));
    float ktv[4]; bcast4(ktv, Ktj);

    // ---- o4 = relu(W4@[S,Kt]+b4); P = relu(W5@[P1,o4]+b5) ----
    float so = b4v;
#pragma unroll
    for (int i = 0; i < 4; ++i) { so = fmaf(w4r[i], S[i], so); so = fmaf(w4r[4+i], ktv[i], so); }
    float o4v[4]; bcast4(o4v, reluf(so));

    float sp = b5v;
#pragma unroll
    for (int i = 0; i < 4; ++i) { sp = fmaf(w5r[i], P1[i], sp); sp = fmaf(w5r[4+i], o4v[i], sp); }
    bcast4(P, reluf(sp));      // carry

    if (lane < 4) out[(size_t)t * 4 + lane] = Ktj;
  };

  // ---- software-pipelined main loop (a loaded one step ahead) ----
  float a0[24], a1[24];
  if (useA) load_a24(a0, Aprec);
  for (int t = 0; t < T; t += 2) {
    if (useA) {
      if (t + 1 < T) load_a24(a1, Aprec + (size_t)(t + 1) * 24);
      if (t + 4 < T) __builtin_prefetch(Aprec + (size_t)(t + 4) * 24, 0, 3);
    } else {
      compute_a(a0, t);
    }
    step(a0, t);
    if (t + 1 < T) {
      if (useA) { if (t + 2 < T) load_a24(a0, Aprec + (size_t)(t + 2) * 24); }
      else      compute_a(a1, t + 1);
      step(a1, t + 1);
    }
  }
}

// ---------------------------------------------------------------------------
extern "C" void kernel_launch(void* const* d_in, const int* in_sizes, int n_in,
                              void* d_out, int out_size, void* d_ws, size_t ws_size,
                              hipStream_t stream) {
  (void)n_in; (void)out_size;
  const float* x    = (const float*)d_in[0];
  const float* W1   = (const float*)d_in[1];
  const float* b1   = (const float*)d_in[2];
  const float* cv1w = (const float*)d_in[3];
  const float* cv1b = (const float*)d_in[4];
  const float* cv2w = (const float*)d_in[5];
  const float* cv2b = (const float*)d_in[6];
  const float* cv3w = (const float*)d_in[7];
  const float* cv3b = (const float*)d_in[8];
  const float* cv4w = (const float*)d_in[9];
  const float* cv4b = (const float*)d_in[10];
  const float* cv5w = (const float*)d_in[11];
  const float* cv5b = (const float*)d_in[12];
  const float* cv6w = (const float*)d_in[13];
  const float* cv6b = (const float*)d_in[14];
  const float* cv7w = (const float*)d_in[15];
  const float* cv7b = (const float*)d_in[16];
  const float* Wih  = (const float*)d_in[17];
  const float* bih  = (const float*)d_in[18];
  const float* bhh  = (const float*)d_in[19];
  const float* W3   = (const float*)d_in[20];
  const float* b3   = (const float*)d_in[21];
  const float* W4   = (const float*)d_in[22];
  const float* b4   = (const float*)d_in[23];
  const float* W5   = (const float*)d_in[24];
  const float* b5   = (const float*)d_in[25];

  const int T = in_sizes[0] / 2;

  float* Abuf = nullptr;
  if (ws_size >= (size_t)T * 24u * sizeof(float)) {
    Abuf = (float*)d_ws;
    const int ntiles  = (T + 15) / 16;        // one wave per 16-row tile
    const int threads = ntiles * 32;
    const int block   = 256;
    const int grid    = (threads + block - 1) / block;
    prep_fc1_wmma<<<grid, block, 0, stream>>>(x, W1, b1, Abuf, T);
  }

  scan_kernel<<<1, 32, 0, stream>>>(x, W1, b1, cv1w, cv1b, cv2w, cv2b,
                                    cv3w, cv3b, cv4w, cv4b, cv5w, cv5b,
                                    cv6w, cv6b, cv7w, cv7b, Wih, bih, bhh,
                                    W3, b3, W4, b4, W5, b5,
                                    Abuf, (float*)d_out, T);
}
